// StreamingAttention_39599598469384
// MI455X (gfx1250) — matrix operand order repaired and checked by reading, not compiled, not run on hardware
//
#include <hip/hip_runtime.h>
#include <hip/hip_bf16.h>
#include <math.h>

// ---------------------------------------------------------------------------
// Problem constants (match reference): B=32, T=4096, D=256, A=128, window=64
// ---------------------------------------------------------------------------
#define B_DIM   32
#define T_DIM   4096
#define D_DIM   256
#define A_DIM   128
#define WIN     64

typedef __attribute__((ext_vector_type(16))) _Float16 v16h;
typedef __attribute__((ext_vector_type(8)))  float    v8f;
typedef __attribute__((ext_vector_type(4)))  float    v4f;

// ---------------------------------------------------------------------------
// Phase 1: g[b*T+t] = sigmoid( Wg . tanh(x[t] . Wp^T + bp) )
// One wave (32 lanes) owns 16 tokens; full A=128 outputs via 8 WMMA N-tiles,
// K=256 consumed as 8 chunks of 32 with v_wmma_f32_16x16x32_f16.
// ---------------------------------------------------------------------------
#define P1_WAVES   4
#define P1_TOKENS  (P1_WAVES * 16)   // 64 tokens per block

__global__ __launch_bounds__(P1_WAVES * 32)
void gate_kernel(const float* __restrict__ x,
                 const float* __restrict__ Wp,
                 const float* __restrict__ bp,
                 const float* __restrict__ Wg,
                 float* __restrict__ g_out)
{
    // Wp as f16, row-major [A][D] in LDS (64 KB; WGP LDS budget is 320 KB).
    __shared__ _Float16 wp_h[A_DIM * D_DIM];

    const int tid = threadIdx.x;
    for (int i = tid; i < A_DIM * D_DIM; i += blockDim.x)
        wp_h[i] = (_Float16)Wp[i];
    __syncthreads();

    const int wave = tid >> 5;
    const int lane = tid & 31;
    const int tok0 = blockIdx.x * P1_TOKENS + wave * 16;

    const int mrow = lane & 15;   // token row this lane feeds into the A frag
    const int kgrp = lane >> 4;   // which K half-group this lane holds

    const float* xrow = x + (size_t)(tok0 + mrow) * D_DIM;

    v8f acc[8] = {};  // 8 N-tiles of the 16x16 f32 accumulator

    for (int kc = 0; kc < 8; ++kc) {
        const int kbase = kc * 32;
        // A fragment (16x32 f16): lanes 0-15 -> K kb..kb+15,
        // lanes 16-31 -> K kb+16..kb+31  (contiguous, matches B)
        v16h afrag;
        const int o0 = kbase + kgrp * 16;
        const int o1 = kbase + kgrp * 16 + 8;
#pragma unroll
        for (int j = 0; j < 8; ++j) afrag[j]     = (_Float16)xrow[o0 + j];
#pragma unroll
        for (int j = 0; j < 8; ++j) afrag[8 + j] = (_Float16)xrow[o1 + j];

#pragma unroll
        for (int nt = 0; nt < 8; ++nt) {
            // B fragment (32x16 f16): lane n = column N, lanes 0-15 hold
            // K=kb..kb+15, lanes 16-31 hold K=kb+16..kb+31, 2 halfs per VGPR.
            // B[k][n] = Wp[a = nt*16+n][d = k]  -> contiguous 16 halfs per lane.
            const _Float16* brow =
                &wp_h[(nt * 16 + mrow) * D_DIM + kbase + kgrp * 16];
            v16h bfrag = *(const v16h*)brow;

            acc[nt] = __builtin_amdgcn_wmma_f32_16x16x32_f16(
                false, afrag, false, bfrag, (short)0, acc[nt], false, false);
        }
    }

    // C/D layout: VGPR r, lanes 0-15 -> (M=r, N=lane); lanes 16-31 -> (M=r+8).
    // Each lane handles one a-column per tile: a = nt*16 + (lane&15).
    float partial[8];
#pragma unroll
    for (int r = 0; r < 8; ++r) partial[r] = 0.f;

#pragma unroll
    for (int nt = 0; nt < 8; ++nt) {
        const int a   = nt * 16 + mrow;
        const float bpv = bp[a];
        const float wgv = Wg[a];
#pragma unroll
        for (int r = 0; r < 8; ++r)
            partial[r] += wgv * tanhf(acc[nt][r] + bpv);
    }

    // Reduce across the 16 N-lanes of each half-wave group (wave32 butterfly).
#pragma unroll
    for (int off = 1; off < 16; off <<= 1) {
#pragma unroll
        for (int r = 0; r < 8; ++r)
            partial[r] += __shfl_xor(partial[r], off, 32);
    }

    if (mrow == 0) {
        const int mbase = kgrp * 8;   // group 0 -> M 0..7, group 1 -> M 8..15
#pragma unroll
        for (int r = 0; r < 8; ++r) {
            const float s = 1.f / (1.f + __expf(-partial[r]));
            g_out[tok0 + mbase + r] = s;
        }
    }
}

// ---------------------------------------------------------------------------
// Phase 2: causal window-64 weighted average via running add/subtract scan.
// x (134 MB) fits in the 192 MB L2 after phase 1, so this phase is mostly
// L2-resident: minimize instructions/byte with float4 (b128) accesses and
// keep x resident by writing `out` with non-temporal stores.
// 64 lanes cover D=256 (4 floats each); 256-thread block = 4 chunk groups.
// ---------------------------------------------------------------------------
#define P2_TC     64    // tokens per 64-lane group
#define P2_GROUPS 4     // 64-lane groups per block

__global__ __launch_bounds__(64 * P2_GROUPS)
void window_kernel(const float* __restrict__ x,
                   const float* __restrict__ g,
                   float* __restrict__ out)
{
    const int lane64 = threadIdx.x & 63;   // which float4 column of D
    const int grp    = threadIdx.x >> 6;   // token-chunk group within block
    const int b      = blockIdx.y;
    const int t0     = (blockIdx.x * P2_GROUPS + grp) * P2_TC;

    const v4f* xb = (const v4f*)(x + (size_t)b * T_DIM * D_DIM);
    v4f*       ob = (v4f*)(out + (size_t)b * T_DIM * D_DIM);
    const float* gb = g + (size_t)b * T_DIM;

    const int DW = D_DIM / 4;   // 64 float4 per token row

    v4f   num = {0.f, 0.f, 0.f, 0.f};
    float den = 0.f;

    // Warm-up: window tail preceding this chunk (L2 hits after phase 1).
    int warm0 = t0 - (WIN - 1);
    if (warm0 < 0) warm0 = 0;
    for (int s = warm0; s < t0; ++s) {
        const float gv = gb[s];
        num += gv * xb[(size_t)s * DW + lane64];
        den += gv;
    }

    for (int t = t0; t < t0 + P2_TC; ++t) {
        const float gv = gb[t];
        num += gv * xb[(size_t)t * DW + lane64];
        den += gv;

        // Output is never re-read: non-temporal store keeps x hot in L2.
        __builtin_nontemporal_store(num / den, &ob[(size_t)t * DW + lane64]);

        const int told = t - (WIN - 1);
        if (told >= 0) {
            const float go = gb[told];
            num -= go * xb[(size_t)told * DW + lane64];
            den -= go;
        }
    }
}

// ---------------------------------------------------------------------------
// Host-side launcher
// ---------------------------------------------------------------------------
extern "C" void kernel_launch(void* const* d_in, const int* in_sizes, int n_in,
                              void* d_out, int out_size, void* d_ws, size_t ws_size,
                              hipStream_t stream)
{
    const float* x  = (const float*)d_in[0];   // [B, T, D]
    const float* Wp = (const float*)d_in[1];   // [A, D]
    const float* bp = (const float*)d_in[2];   // [A]
    const float* Wg = (const float*)d_in[3];   // [1, A]
    float* out = (float*)d_out;                // [B, T, D]
    float* g   = (float*)d_ws;                 // [B*T] gate scratch (512 KB)

    (void)in_sizes; (void)n_in; (void)out_size; (void)ws_size;

    // Phase 1: gates via WMMA. 131072 tokens / 64 per block = 2048 blocks.
    gate_kernel<<<dim3((B_DIM * T_DIM) / P1_TOKENS), P1_WAVES * 32, 0, stream>>>(
        x, Wp, bp, Wg, g);

    // Phase 2: sliding-window normalize.
    // grid.x = 4096 / (64*4) = 16 chunk-blocks, grid.y = 32 batches.
    window_kernel<<<dim3(T_DIM / (P2_TC * P2_GROUPS), B_DIM),
                    64 * P2_GROUPS, 0, stream>>>(x, g, out);
}
